// StyleGan2Gen_37692632989928
// MI455X (gfx1250) — matrix operand order repaired and checked by reading, not compiled, not run on hardware
//
#include <hip/hip_runtime.h>
#include <hip/hip_bf16.h>

// ---------------------------------------------------------------------------
// StyleGAN2-like generator forward for MI455X (gfx1250, wave32, WMMA).
// Convs = implicit GEMM on v_wmma_f32_16x16x32_bf16 with NHWC(Cpad) layout so
// every A/B fragment is two contiguous 16B loads; the K loop walks (ky,(kx,ci))
// with pure pointer increments (no per-chunk index re-derivation).
// ---------------------------------------------------------------------------

typedef __bf16 bf16_t;
typedef __attribute__((ext_vector_type(16))) __bf16 v16bf;
typedef __attribute__((ext_vector_type(8)))  __bf16 v8bf;
typedef __attribute__((ext_vector_type(8)))  float v8f;

#define CDIV(a, b) (((a) + (b) - 1) / (b))

// ----------------------------- device helpers ------------------------------

__device__ __forceinline__ int reflect_idx(int i, int n) {
  // jnp.pad mode='reflect' (no edge repeat); pad < n always holds here.
  if (i < 0) i = -i;
  if (i >= n) i = 2 * n - 2 - i;
  return i;
}

__device__ __forceinline__ v16bf cat8(v8bf lo, v8bf hi) {
  return __builtin_shufflevector(lo, hi, 0, 1, 2, 3, 4, 5, 6, 7,
                                 8, 9, 10, 11, 12, 13, 14, 15);
}

// ---------------- pad (+channel scale) -> NHWC bf16, Cpad channels ---------
// out: (B, Hp, Wp, Cpad) bf16; channels >= Cin are zero. mode: 1=reflect,0=zero.
// effective scale = scale[b*Cin+c] + 1 (folds StyleGAN (style+1) modulation).
__global__ __launch_bounds__(256) void k_pad_nhwc(
    const float* __restrict__ x, bf16_t* __restrict__ y,
    int Cin, int Cpad, int H, int W, int pad, int reflect,
    const float* __restrict__ scale, long total) {
  long idx = (long)blockIdx.x * blockDim.x + threadIdx.x;
  if (idx >= total) return;
  int Wp = W + 2 * pad, Hp = H + 2 * pad;
  int c = (int)(idx % Cpad); long t = idx / Cpad;
  int xx = (int)(t % Wp); t /= Wp;
  int yy = (int)(t % Hp); int b = (int)(t / Hp);
  float v = 0.0f;
  if (c < Cin) {
    int sx = xx - pad, sy = yy - pad;
    bool in = true;
    if (reflect) {
      sx = reflect_idx(sx, W);
      sy = reflect_idx(sy, H);
    } else {
      in = (sx >= 0 && sx < W && sy >= 0 && sy < H);
    }
    if (in) {
      v = x[((long)(b * Cin + c) * H + sy) * W + sx];
      if (scale) v *= (scale[b * Cin + c] + 1.0f);
    }
  }
  y[idx] = (bf16_t)v;
}

// --------- weight OIHW f32 -> [Cout][Kpad] bf16, K=(ky,kx,ci), zero-fill ---
__global__ __launch_bounds__(256) void k_wt_nhwc(
    const float* __restrict__ w, bf16_t* __restrict__ out,
    int Cin, int Cpad, int KH, int KW, long total) {
  long idx = (long)blockIdx.x * blockDim.x + threadIdx.x;
  if (idx >= total) return;
  long Kpad = (long)KH * KW * Cpad;
  int k = (int)(idx % Kpad);
  int n = (int)(idx / Kpad);
  int ci = k % Cpad;
  int r = k / Cpad;
  int ky = r / KW, kx = r - ky * KW;
  float v = 0.0f;
  if (ci < Cin) v = w[(((long)n * Cin + ci) * KH + ky) * KW + kx];
  out[idx] = (bf16_t)v;
}

// -------------------- implicit-GEMM conv via bf16 WMMA ---------------------
// Xp: (B,Hp,Wp,Cpad) bf16. Wt: (Cout,Kpad) bf16. Y: (B,Cout,OH,OW) f32.
// One wave computes 64(M) x 32(N); K chunked by 32. The weight pointer walks
// contiguously across all of K; A row pointers advance by 32 within a ky row.
__global__ __launch_bounds__(32) void k_wmma_conv(
    const bf16_t* __restrict__ Xp, const bf16_t* __restrict__ Wt,
    const float* __restrict__ bias, float* __restrict__ Y,
    int Cpad, int Hp, int Wp, int Cout, int KH, int KW, int stride,
    int OH, int OW) {
  const int lane = threadIdx.x;
  const int m0 = blockIdx.x * 64;
  const int n0 = blockIdx.y * 32;
  const int HW = OH * OW;
  const long Kpad = (long)KH * KW * Cpad;
  const long WpC = (long)Wp * Cpad;
  const int rowK = KW * Cpad;  // K elements per ky row (multiple of 32)

  // A-fragment row bases (4 M sub-tiles of 16)
  long abase[4];
#pragma unroll
  for (int s = 0; s < 4; ++s) {
    int m = m0 + s * 16 + (lane & 15);
    int b = m / HW; int r = m - b * HW;
    int oy = r / OW; int ox = r - oy * OW;
    abase[s] = (((long)b * Hp + oy * stride) * Wp + ox * stride) * (long)Cpad;
  }
  const int aOff = (lane < 16) ? 0 : 8;  // A lane-group K offset

  // B pointers: column n0+(lane&15) [+16], lane-group K offset; contiguous in K.
  const bf16_t* wp0 = Wt + (long)(n0 + (lane & 15)) * Kpad + ((lane >> 4) * 8);
  const bf16_t* wp1 = wp0 + 16 * Kpad;

  v8f z = {0.f, 0.f, 0.f, 0.f, 0.f, 0.f, 0.f, 0.f};
  v8f acc[4][2] = {{z, z}, {z, z}, {z, z}, {z, z}};

  for (int ky = 0; ky < KH; ++ky) {
    const bf16_t* a0 = Xp + abase[0] + (long)ky * WpC + aOff;
    const bf16_t* a1 = Xp + abase[1] + (long)ky * WpC + aOff;
    const bf16_t* a2 = Xp + abase[2] + (long)ky * WpC + aOff;
    const bf16_t* a3 = Xp + abase[3] + (long)ky * WpC + aOff;
    for (int kc = 0; kc < rowK; kc += 32) {
      v16bf b0 = cat8(*(const v8bf*)wp0, *(const v8bf*)(wp0 + 16));
      v16bf b1 = cat8(*(const v8bf*)wp1, *(const v8bf*)(wp1 + 16));
      v16bf f0 = cat8(*(const v8bf*)a0, *(const v8bf*)(a0 + 16));
      v16bf f1 = cat8(*(const v8bf*)a1, *(const v8bf*)(a1 + 16));
      v16bf f2 = cat8(*(const v8bf*)a2, *(const v8bf*)(a2 + 16));
      v16bf f3 = cat8(*(const v8bf*)a3, *(const v8bf*)(a3 + 16));
      acc[0][0] = __builtin_amdgcn_wmma_f32_16x16x32_bf16(
          false, f0, false, b0, (short)0, acc[0][0], false, false);
      acc[0][1] = __builtin_amdgcn_wmma_f32_16x16x32_bf16(
          false, f0, false, b1, (short)0, acc[0][1], false, false);
      acc[1][0] = __builtin_amdgcn_wmma_f32_16x16x32_bf16(
          false, f1, false, b0, (short)0, acc[1][0], false, false);
      acc[1][1] = __builtin_amdgcn_wmma_f32_16x16x32_bf16(
          false, f1, false, b1, (short)0, acc[1][1], false, false);
      acc[2][0] = __builtin_amdgcn_wmma_f32_16x16x32_bf16(
          false, f2, false, b0, (short)0, acc[2][0], false, false);
      acc[2][1] = __builtin_amdgcn_wmma_f32_16x16x32_bf16(
          false, f2, false, b1, (short)0, acc[2][1], false, false);
      acc[3][0] = __builtin_amdgcn_wmma_f32_16x16x32_bf16(
          false, f3, false, b0, (short)0, acc[3][0], false, false);
      acc[3][1] = __builtin_amdgcn_wmma_f32_16x16x32_bf16(
          false, f3, false, b1, (short)0, acc[3][1], false, false);
      a0 += 32; a1 += 32; a2 += 32; a3 += 32;
      wp0 += 32; wp1 += 32;
    }
  }

  // ---- writeback: C layout lane-group g -> M=8g+v, N=lane&15 ----
#pragma unroll
  for (int s = 0; s < 4; ++s) {
#pragma unroll
    for (int t = 0; t < 2; ++t) {
      int n = n0 + t * 16 + (lane & 15);
      float bv = bias ? bias[n] : 0.0f;
#pragma unroll
      for (int v = 0; v < 8; ++v) {
        int m = m0 + s * 16 + ((lane >> 4) * 8) + v;
        int b = m / HW; int r = m - b * HW;
        int oy = r / OW; int ox = r - oy * OW;
        Y[(((long)b * Cout + n) * OH + oy) * OW + ox] = acc[s][t][v] + bv;
      }
    }
  }
}

// ------------------------- instance norm (+ReLU) ---------------------------
__global__ __launch_bounds__(256) void k_instnorm_relu(
    float* __restrict__ x, int HW, float eps, int do_relu) {
  __shared__ float ssum[256], ssq[256];
  __shared__ float s_mu, s_rstd;
  const long base = (long)blockIdx.x * HW;
  const int tid = threadIdx.x;
  float s = 0.f, q = 0.f;
  for (int i = tid; i < HW; i += 256) {
    float v = x[base + i];
    s += v; q += v * v;
  }
  ssum[tid] = s; ssq[tid] = q;
  __syncthreads();
  for (int step = 128; step > 0; step >>= 1) {
    if (tid < step) { ssum[tid] += ssum[tid + step]; ssq[tid] += ssq[tid + step]; }
    __syncthreads();
  }
  if (tid == 0) {
    float mu = ssum[0] / (float)HW;
    float var = ssq[0] / (float)HW - mu * mu;
    s_mu = mu;
    s_rstd = rsqrtf(var + eps);
  }
  __syncthreads();
  float mu = s_mu, rstd = s_rstd;
  for (int i = tid; i < HW; i += 256) {
    float v = (x[base + i] - mu) * rstd;
    if (do_relu) v = v > 0.f ? v : 0.f;
    x[base + i] = v;
  }
}

// --------------------------- per-(b,c) mean --------------------------------
__global__ __launch_bounds__(256) void k_chan_mean(
    const float* __restrict__ x, float* __restrict__ out, int HW) {
  __shared__ float ssum[256];
  const long base = (long)blockIdx.x * HW;
  const int tid = threadIdx.x;
  float s = 0.f;
  for (int i = tid; i < HW; i += 256) s += x[base + i];
  ssum[tid] = s;
  __syncthreads();
  for (int step = 128; step > 0; step >>= 1) {
    if (tid < step) ssum[tid] += ssum[tid + step];
    __syncthreads();
  }
  if (tid == 0) out[blockIdx.x] = ssum[0] / (float)HW;
}

// ------------------------------ small linear -------------------------------
__global__ __launch_bounds__(128) void k_linear(
    const float* __restrict__ xin, const float* __restrict__ Wm,
    const float* __restrict__ bias, float* __restrict__ out,
    int I, int O, int total) {
  int idx = blockIdx.x * blockDim.x + threadIdx.x;
  if (idx >= total) return;
  int b = idx / O, o = idx % O;
  float s = bias ? bias[o] : 0.f;
  const float* xi = xin + (long)b * I;
  const float* wr = Wm + (long)o * I;
  for (int i = 0; i < I; ++i) s += xi[i] * wr[i];
  out[idx] = s;
}

// -------------------- sum over kernel window of w^2 ------------------------
__global__ __launch_bounds__(256) void k_wsqsum(
    const float* __restrict__ w, float* __restrict__ wsq, int KK, int total) {
  int idx = blockIdx.x * blockDim.x + threadIdx.x;
  if (idx >= total) return;
  const float* p = w + (long)idx * KK;
  float s = 0.f;
  for (int k = 0; k < KK; ++k) s += p[k] * p[k];
  wsq[idx] = s;
}

// --------------------------- demod coefficients ----------------------------
// d[b,o] = rsqrt( sum_i wsq[o*I+i]*(s1[b*I+i]+1)^2 + 1e-8 )
__global__ __launch_bounds__(128) void k_demod(
    const float* __restrict__ wsq, const float* __restrict__ s1,
    float* __restrict__ d, int O, int I, int total) {
  int idx = blockIdx.x * blockDim.x + threadIdx.x;
  if (idx >= total) return;
  int b = idx / O, o = idx % O;
  const float* wr = wsq + (long)o * I;
  const float* sr = s1 + (long)b * I;
  float acc = 0.f;
  for (int i = 0; i < I; ++i) {
    float m = sr[i] + 1.0f;
    acc += wr[i] * m * m;
  }
  d[idx] = rsqrtf(acc + 1e-8f);
}

// ---------------- epilogue: demod scale + noise + leaky --------------------
// x[b,c,y,xx] = leaky(x*d[b,c] + noise[b,xx,y,0]*tw[c] + tb[c]); noise 256x256.
__global__ __launch_bounds__(256) void k_epilogue(
    float* __restrict__ x, const float* __restrict__ d,
    const float* __restrict__ noise, const float* __restrict__ tw,
    const float* __restrict__ tb, int C, int H, int W, long total) {
  long idx = (long)blockIdx.x * blockDim.x + threadIdx.x;
  if (idx >= total) return;
  int xx = (int)(idx % W); long t = idx / W;
  int y = (int)(t % H); t /= H;
  int c = (int)(t % C); int b = (int)(t / C);
  float nz = noise[((long)b * 256 + xx) * 256 + y];
  float v = x[idx] * d[b * C + c] + nz * tw[c] + tb[c];
  x[idx] = v >= 0.f ? v : 0.2f * v;
}

// ---------------------- bilinear 2x upsample (half-pixel) ------------------
__global__ __launch_bounds__(256) void k_upsample2x(
    const float* __restrict__ xin, float* __restrict__ out,
    int C, int H, int W, long total) {
  long idx = (long)blockIdx.x * blockDim.x + threadIdx.x;
  if (idx >= total) return;
  int W2 = 2 * W, H2 = 2 * H;
  int ox = (int)(idx % W2); long t = idx / W2;
  int oy = (int)(t % H2); t /= H2;
  int c = (int)(t % C); int b = (int)(t / C);
  float fx = (ox + 0.5f) * 0.5f - 0.5f;
  float fy = (oy + 0.5f) * 0.5f - 0.5f;
  int x0 = (int)floorf(fx); float wx = fx - (float)x0;
  int y0 = (int)floorf(fy); float wy = fy - (float)y0;
  int x0c = min(max(x0, 0), W - 1), x1c = min(max(x0 + 1, 0), W - 1);
  int y0c = min(max(y0, 0), H - 1), y1c = min(max(y0 + 1, 0), H - 1);
  const float* p = xin + (long)(b * C + c) * H * W;
  float v00 = p[(long)y0c * W + x0c], v01 = p[(long)y0c * W + x1c];
  float v10 = p[(long)y1c * W + x0c], v11 = p[(long)y1c * W + x1c];
  out[idx] = (1.f - wy) * ((1.f - wx) * v00 + wx * v01) +
             wy * ((1.f - wx) * v10 + wx * v11);
}

// ---------------- depthwise 3x3 blur [1,2,1]^2/16, reflect pad -------------
__global__ __launch_bounds__(256) void k_blur3(
    const float* __restrict__ xin, float* __restrict__ out,
    int C, int H, int W, long total) {
  long idx = (long)blockIdx.x * blockDim.x + threadIdx.x;
  if (idx >= total) return;
  int xx = (int)(idx % W); long t = idx / W;
  int y = (int)(t % H); t /= H;
  int c = (int)(t % C); int b = (int)(t / C);
  const float* p = xin + (long)(b * C + c) * H * W;
  const float wgt[3] = {1.f, 2.f, 1.f};
  float s = 0.f;
#pragma unroll
  for (int dy = 0; dy < 3; ++dy) {
    int yy = reflect_idx(y + dy - 1, H);
#pragma unroll
    for (int dx = 0; dx < 3; ++dx) {
      int xc = reflect_idx(xx + dx - 1, W);
      s += p[(long)yy * W + xc] * wgt[dy] * wgt[dx];
    }
  }
  out[idx] = s * (1.f / 16.f);
}

// ------------- modulated 1x1 RGB conv (no demod), optional add prev --------
__global__ __launch_bounds__(256) void k_rgb1x1(
    const float* __restrict__ x, const float* __restrict__ w,
    const float* __restrict__ rs, const float* __restrict__ prev,
    float* __restrict__ out, int Cin, int H, int W, long total) {
  long idx = (long)blockIdx.x * blockDim.x + threadIdx.x;
  if (idx >= total) return;
  int xx = (int)(idx % W); long t = idx / W;
  int y = (int)(t % H); t /= H;
  int o = (int)(t % 3); int b = (int)(t / 3);
  const float* xb = x + ((long)b * Cin * H + y) * W + xx;
  const long hw = (long)H * W;
  float s = 0.f;
  for (int i = 0; i < Cin; ++i)
    s += xb[(long)i * hw] * w[o * Cin + i] * (rs[b * Cin + i] + 1.0f);
  if (prev) s += prev[idx];
  out[idx] = s;
}

// ---------------------------------- tanh -----------------------------------
__global__ __launch_bounds__(256) void k_tanh(
    const float* __restrict__ xin, float* __restrict__ out, long total) {
  long idx = (long)blockIdx.x * blockDim.x + threadIdx.x;
  if (idx >= total) return;
  out[idx] = tanhf(xin[idx]);
}

// ===========================================================================
//                                host side
// ===========================================================================

extern "C" void kernel_launch(void* const* d_in, const int* in_sizes, int n_in,
                              void* d_out, int out_size, void* d_ws, size_t ws_size,
                              hipStream_t stream) {
  (void)in_sizes; (void)n_in; (void)out_size; (void)ws_size;
  const int B = 4;

  const float* img_A = (const float*)d_in[0];    // (4,3,256,256)
  const float* sem_B = (const float*)d_in[1];    // (4,24,3,256,256) == (4,72,256,256)
  const float* noise = (const float*)d_in[2];    // (4,256,256,1)
  const float* ecw[3] = {(const float*)d_in[3], (const float*)d_in[5], (const float*)d_in[7]};
  const float* ecb[3] = {(const float*)d_in[4], (const float*)d_in[6], (const float*)d_in[8]};
  const float* esw[5] = {(const float*)d_in[9], (const float*)d_in[11], (const float*)d_in[13],
                         (const float*)d_in[15], (const float*)d_in[17]};
  const float* esb[5] = {(const float*)d_in[10], (const float*)d_in[12], (const float*)d_in[14],
                         (const float*)d_in[16], (const float*)d_in[18]};
  const float* pool_W = (const float*)d_in[19];  // (2048,1024,1,1)
  const float* pool_b = (const float*)d_in[20];
  const float* init_w = (const float*)d_in[21];  // (128,128,3,3)
  const float* init_b = (const float*)d_in[22];

  // ---- workspace bump allocator ----
  char* base = (char*)d_ws;
  size_t off = 0;
  auto alloc = [&](size_t bytes) -> void* {
    size_t a = (off + 255) & ~(size_t)255;
    off = a + bytes;
    return (void*)(base + a);
  };

  // ---- conv launcher: NHWC pad+convert, weight convert, WMMA GEMM ----
  auto run_conv = [&](const float* x, const float* w, const float* bias, float* y,
                      int Cin, int H, int W, int Cout, int Kk, int stride, int pad,
                      int reflect, const float* scale) {
    size_t mark = off;
    int Hp = H + 2 * pad, Wp = W + 2 * pad;
    int OH = (Hp - Kk) / stride + 1, OW = (Wp - Kk) / stride + 1;
    int Cpad = (Cin + 31) & ~31;
    long Kpad = (long)Kk * Kk * Cpad;
    long ptot = (long)B * Hp * Wp * Cpad;
    bf16_t* xp = (bf16_t*)alloc((size_t)ptot * sizeof(bf16_t));
    long wtot = Kpad * Cout;
    bf16_t* wt = (bf16_t*)alloc((size_t)wtot * sizeof(bf16_t));
    k_pad_nhwc<<<(unsigned)CDIV(ptot, 256), 256, 0, stream>>>(
        x, xp, Cin, Cpad, H, W, pad, reflect, scale, ptot);
    k_wt_nhwc<<<(unsigned)CDIV(wtot, 256), 256, 0, stream>>>(w, wt, Cin, Cpad, Kk, Kk, wtot);
    dim3 grid((unsigned)((B * OH * OW) / 64), (unsigned)(Cout / 32));
    k_wmma_conv<<<grid, 32, 0, stream>>>(xp, wt, bias, y, Cpad, Hp, Wp, Cout, Kk, Kk,
                                         stride, OH, OW);
    off = mark;  // stream-ordered reuse of pad/weight temps is safe
  };

  auto conv_in_relu = [&](const float* x, float* y, const float* w, const float* b,
                          int Cin, int H, int W, int Cout, int Kk, int stride, int pad) {
    run_conv(x, w, b, y, Cin, H, W, Cout, Kk, stride, pad, /*reflect=*/1, nullptr);
    int OH = (H + 2 * pad - Kk) / stride + 1;
    int OW = (W + 2 * pad - Kk) / stride + 1;
    k_instnorm_relu<<<(unsigned)(B * Cout), 256, 0, stream>>>(y, OH * OW, 1e-5f, 1);
  };

  // ---- persistent small buffers ----
  float* style = (float*)alloc((size_t)B * 2048 * sizeof(float));
  float* smean = (float*)alloc((size_t)B * 1024 * sizeof(float));
  float* c2 = (float*)alloc((size_t)B * 128 * 64 * 64 * sizeof(float));
  float* x0 = (float*)alloc((size_t)B * 128 * 64 * 64 * sizeof(float));
  size_t phase = off;

  // ============================ style encoder ==============================
  {
    float* s0 = (float*)alloc((size_t)B * 64 * 256 * 256 * sizeof(float));
    conv_in_relu(sem_B, s0, esw[0], esb[0], 72, 256, 256, 64, 7, 1, 3);
    float* s1 = (float*)alloc((size_t)B * 128 * 128 * 128 * sizeof(float));
    conv_in_relu(s0, s1, esw[1], esb[1], 64, 256, 256, 128, 4, 2, 1);
    float* s2 = (float*)alloc((size_t)B * 256 * 64 * 64 * sizeof(float));
    conv_in_relu(s1, s2, esw[2], esb[2], 128, 128, 128, 256, 4, 2, 1);
    float* s3 = (float*)alloc((size_t)B * 512 * 32 * 32 * sizeof(float));
    conv_in_relu(s2, s3, esw[3], esb[3], 256, 64, 64, 512, 4, 2, 1);
    float* s4 = (float*)alloc((size_t)B * 1024 * 16 * 16 * sizeof(float));
    conv_in_relu(s3, s4, esw[4], esb[4], 512, 32, 32, 1024, 4, 2, 1);
    k_chan_mean<<<(unsigned)(B * 1024), 256, 0, stream>>>(s4, smean, 256);
    int tot = B * 2048;
    k_linear<<<(unsigned)CDIV(tot, 128), 128, 0, stream>>>(smean, pool_W, pool_b, style,
                                                           1024, 2048, tot);
    off = phase;
  }

  // ========================== content encoder =============================
  {
    float* c0 = (float*)alloc((size_t)B * 32 * 256 * 256 * sizeof(float));
    conv_in_relu(img_A, c0, ecw[0], ecb[0], 3, 256, 256, 32, 7, 1, 3);
    float* c1 = (float*)alloc((size_t)B * 64 * 128 * 128 * sizeof(float));
    conv_in_relu(c0, c1, ecw[1], ecb[1], 32, 256, 256, 64, 4, 2, 1);
    conv_in_relu(c1, c2, ecw[2], ecb[2], 64, 128, 128, 128, 4, 2, 1);
    // initial conv (zero padding, with bias, no norm)
    run_conv(c2, init_w, init_b, x0, 128, 64, 64, 128, 3, 1, 1, /*reflect=*/0, nullptr);
    off = phase;
  }

  // ============================ generator =================================
  auto run_block = [&](const float* xin, int Cin, int f, int H, int W, int pb,
                       const float* rgb_prev, float** rgb_out, float** x_out) {
    const float* ts1_w = (const float*)d_in[pb + 0];
    const float* ts1_b = (const float*)d_in[pb + 1];
    const float* tn1_w = (const float*)d_in[pb + 2];
    const float* tn1_b = (const float*)d_in[pb + 3];
    const float* conv1_w = (const float*)d_in[pb + 4];
    const float* ts2_w = (const float*)d_in[pb + 5];
    const float* ts2_b = (const float*)d_in[pb + 6];
    const float* tn2_w = (const float*)d_in[pb + 7];
    const float* tn2_b = (const float*)d_in[pb + 8];
    const float* conv2_w = (const float*)d_in[pb + 9];
    const float* rgb_s_w = (const float*)d_in[pb + 10];
    const float* rgb_s_b = (const float*)d_in[pb + 11];
    const float* rgb_conv_w = (const float*)d_in[pb + 12];
    long hw = (long)H * W;

    // conv1: modulate input by (s1+1), shared-weight conv, demod in epilogue
    float* s1 = (float*)alloc((size_t)B * Cin * sizeof(float));
    k_linear<<<(unsigned)CDIV(B * Cin, 128), 128, 0, stream>>>(style, ts1_w, ts1_b, s1,
                                                               2048, Cin, B * Cin);
    float* wsq1 = (float*)alloc((size_t)f * Cin * sizeof(float));
    k_wsqsum<<<(unsigned)CDIV(f * Cin, 256), 256, 0, stream>>>(conv1_w, wsq1, 9, f * Cin);
    float* dm1 = (float*)alloc((size_t)B * f * sizeof(float));
    k_demod<<<(unsigned)CDIV(B * f, 128), 128, 0, stream>>>(wsq1, s1, dm1, f, Cin, B * f);
    float* t1 = (float*)alloc((size_t)B * f * hw * sizeof(float));
    run_conv(xin, conv1_w, nullptr, t1, Cin, H, W, f, 3, 1, 1, 0, s1);
    long tot1 = (long)B * f * hw;
    k_epilogue<<<(unsigned)CDIV(tot1, 256), 256, 0, stream>>>(t1, dm1, noise, tn1_w, tn1_b,
                                                              f, H, W, tot1);
    // conv2
    float* s2 = (float*)alloc((size_t)B * f * sizeof(float));
    k_linear<<<(unsigned)CDIV(B * f, 128), 128, 0, stream>>>(style, ts2_w, ts2_b, s2,
                                                             2048, f, B * f);
    float* wsq2 = (float*)alloc((size_t)f * f * sizeof(float));
    k_wsqsum<<<(unsigned)CDIV(f * f, 256), 256, 0, stream>>>(conv2_w, wsq2, 9, f * f);
    float* dm2 = (float*)alloc((size_t)B * f * sizeof(float));
    k_demod<<<(unsigned)CDIV(B * f, 128), 128, 0, stream>>>(wsq2, s2, dm2, f, f, B * f);
    float* t2 = (float*)alloc((size_t)B * f * hw * sizeof(float));
    run_conv(t1, conv2_w, nullptr, t2, f, H, W, f, 3, 1, 1, 0, s2);
    k_epilogue<<<(unsigned)CDIV(tot1, 256), 256, 0, stream>>>(t2, dm2, noise, tn2_w, tn2_b,
                                                              f, H, W, tot1);
    // rgb (1x1 modulated, no demod) + optional residual add
    float* rs = (float*)alloc((size_t)B * f * sizeof(float));
    k_linear<<<(unsigned)CDIV(B * f, 128), 128, 0, stream>>>(style, rgb_s_w, rgb_s_b, rs,
                                                             2048, f, B * f);
    float* rgb = (float*)alloc((size_t)B * 3 * hw * sizeof(float));
    long rtot = (long)B * 3 * hw;
    k_rgb1x1<<<(unsigned)CDIV(rtot, 256), 256, 0, stream>>>(t2, rgb_conv_w, rs, rgb_prev,
                                                            rgb, f, H, W, rtot);
    *x_out = t2;
    *rgb_out = rgb;
  };

  auto up_blur_rgb = [&](const float* rgb, int H, int W) -> float* {
    long tot = (long)B * 3 * 4 * H * W;
    float* ru = (float*)alloc((size_t)tot * sizeof(float));
    k_upsample2x<<<(unsigned)CDIV(tot, 256), 256, 0, stream>>>(rgb, ru, 3, H, W, tot);
    float* rb = (float*)alloc((size_t)tot * sizeof(float));
    k_blur3<<<(unsigned)CDIV(tot, 256), 256, 0, stream>>>(ru, rb, 3, 2 * H, 2 * W, tot);
    return rb;
  };

  {
    // block 0: no upsample, upsample_rgb=True  (128 -> 128 @ 64x64)
    float *rgb0, *x1;
    run_block(x0, 128, 128, 64, 64, 23, nullptr, &rgb0, &x1);
    float* rgbp = up_blur_rgb(rgb0, 64, 64);  // -> (B,3,128,128)

    // block 1: upsample, upsample_rgb=True   (128 -> 64 @ 128x128)
    long xt1 = (long)B * 128 * 128 * 128;
    float* xu1 = (float*)alloc((size_t)xt1 * sizeof(float));
    k_upsample2x<<<(unsigned)CDIV(xt1, 256), 256, 0, stream>>>(x1, xu1, 128, 64, 64, xt1);
    float *rgb1, *x2;
    run_block(xu1, 128, 64, 128, 128, 36, rgbp, &rgb1, &x2);
    rgbp = up_blur_rgb(rgb1, 128, 128);  // -> (B,3,256,256)

    // block 2: upsample, no rgb upsample    (64 -> 32 @ 256x256)
    long xt2 = (long)B * 64 * 256 * 256;
    float* xu2 = (float*)alloc((size_t)xt2 * sizeof(float));
    k_upsample2x<<<(unsigned)CDIV(xt2, 256), 256, 0, stream>>>(x2, xu2, 64, 128, 128, xt2);
    float *rgb2, *x3;
    run_block(xu2, 64, 32, 256, 256, 49, rgbp, &rgb2, &x3);
    (void)x3;

    long ftot = (long)B * 3 * 256 * 256;
    k_tanh<<<(unsigned)CDIV(ftot, 256), 256, 0, stream>>>(rgb2, (float*)d_out, ftot);
  }
}
